// T5LayerSelfAttention_73014444032600
// MI455X (gfx1250) — matrix operand order repaired and checked by reading, not compiled, not run on hardware
//
#include <hip/hip_runtime.h>

typedef __attribute__((ext_vector_type(8)))  float  v8f;
typedef __attribute__((ext_vector_type(16))) __bf16 v16bf;
typedef __attribute__((ext_vector_type(8)))  __bf16 v8bf;
typedef __attribute__((ext_vector_type(4)))  int    v4i;

static constexpr int B_ = 2, S_ = 2048, D_ = 1024, H_ = 16, DK_ = 64;
static constexpr int HD_ = H_ * DK_;        // 1024
static constexpr int NBUCK = 32, MAXDIST = 128;
static constexpr int RELSPAN = 2 * S_ - 1;  // 4095

#if defined(__has_builtin)
#if __has_builtin(__builtin_amdgcn_global_load_async_to_lds_b128)
#define HAVE_ASYNC_LDS 1
#endif
#endif

#ifdef HAVE_ASYNC_LDS
typedef __attribute__((address_space(1))) v4i* gv4ip;   // global 128-bit chunk
typedef __attribute__((address_space(3))) v4i* sv4ip;   // LDS 128-bit chunk
__device__ __forceinline__ void async_copy16(const __bf16* g, __bf16* l) {
  __builtin_amdgcn_global_load_async_to_lds_b128((gv4ip)(void*)g, (sv4ip)(void*)l,
                                                 0, 0);
}
__device__ __forceinline__ void wait_async0() {
  asm volatile("s_wait_asynccnt 0x0" ::: "memory");
}
#endif

__device__ __forceinline__ v16bf cat8(v8bf lo, v8bf hi) {
  return __builtin_shufflevector(lo, hi, 0,1,2,3,4,5,6,7,8,9,10,11,12,13,14,15);
}
__device__ __forceinline__ float rmax16(float v) {
  #pragma unroll
  for (int o = 1; o < 16; o <<= 1) v = fmaxf(v, __shfl_xor(v, o, 16));
  return v;
}
__device__ __forceinline__ float rsum16(float v) {
  #pragma unroll
  for (int o = 1; o < 16; o <<= 1) v += __shfl_xor(v, o, 16);
  return v;
}
__device__ __forceinline__ v8f wmma_bf16(v16bf a, v16bf b, v8f c) {
  return __builtin_amdgcn_wmma_f32_16x16x32_bf16(false, a, false, b, (short)0, c,
                                                 false, false);
}

// ---------------- weight transpose + f32 -> bf16 -----------------------------
__global__ void k_wT(const float* __restrict__ W, __bf16* __restrict__ WT) {
  int i = blockIdx.x * 256 + threadIdx.x;   // 1024*1024 total
  int r = i >> 10, c = i & 1023;
  WT[(size_t)c * 1024 + r] = (__bf16)W[i];
}

// ---------------- T5 relative-position bias LUT ------------------------------
__global__ void k_bias_lut(const float* __restrict__ rel_bias, float* __restrict__ lut) {
  int i = blockIdx.x * 256 + threadIdx.x;
  if (i >= H_ * RELSPAN) return;
  int h = i / RELSPAN, idx = i - h * RELSPAN;
  int rel = idx - (S_ - 1);
  const int nb = NBUCK / 2;       // 16
  const int max_exact = nb / 2;   // 8
  int ret = (rel > 0) ? nb : 0;
  int n = rel < 0 ? -rel : rel;
  int bucket;
  if (n < max_exact) {
    bucket = ret + n;
  } else {
    float v = logf((float)n / (float)max_exact) /
              logf((float)MAXDIST / (float)max_exact) * (float)(nb - max_exact);
    int vi = max_exact + (int)v;
    if (vi > nb - 1) vi = nb - 1;
    bucket = ret + vi;
  }
  lut[i] = rel_bias[bucket * H_ + h];
}

// ---------------- LayerNorm -> bf16 ------------------------------------------
__global__ void k_layernorm(const float* __restrict__ x, const float* __restrict__ gm,
                            const float* __restrict__ bt, __bf16* __restrict__ out) {
  const int row = blockIdx.x;
  const int tid = threadIdx.x;               // 256 threads
  const float* xr = x + (size_t)row * D_;
  float v[4];
  float s = 0.f;
  #pragma unroll
  for (int i = 0; i < 4; ++i) { v[i] = xr[tid + i * 256]; s += v[i]; }
  __shared__ float red[8];
  #pragma unroll
  for (int o = 1; o < 32; o <<= 1) s += __shfl_xor(s, o, 32);
  const int wave = tid >> 5, lane = tid & 31;
  if (lane == 0) red[wave] = s;
  __syncthreads();
  float tot = 0.f;
  #pragma unroll
  for (int w = 0; w < 8; ++w) tot += red[w];
  const float mu = tot * (1.f / D_);
  __syncthreads();
  float sq = 0.f;
  #pragma unroll
  for (int i = 0; i < 4; ++i) { float d = v[i] - mu; sq += d * d; }
  #pragma unroll
  for (int o = 1; o < 32; o <<= 1) sq += __shfl_xor(sq, o, 32);
  if (lane == 0) red[wave] = sq;
  __syncthreads();
  float tsq = 0.f;
  #pragma unroll
  for (int w = 0; w < 8; ++w) tsq += red[w];
  const float rs = rsqrtf(tsq * (1.f / D_) + 1e-6f);
  #pragma unroll
  for (int i = 0; i < 4; ++i) {
    int c = tid + i * 256;
    out[(size_t)row * D_ + c] = (__bf16)((v[i] - mu) * rs * gm[c] + bt[c]);
  }
}

// ---------------- WMMA GEMM: C[M x 1024] = A[M x 1024] * BT^T ----------------
// Block tile 128x128, 8 waves, each wave 32x64 (8 WMMA / k-step).
// Double-buffered LDS; tiles filled by async global->LDS DMA when available.
// mode 0: bf16 out, natural [m][n]
// mode 1: bf16 out, transposed per batch: out[(b*1024+n)*2048 + s], m=b*2048+s
// mode 2: f32 out = resid + acc, natural
__global__ void k_gemm(const __bf16* __restrict__ A, const __bf16* __restrict__ BT,
                       int mode, __bf16* __restrict__ outb, float* __restrict__ outf,
                       const float* __restrict__ resid) {
  __shared__ alignas(32) __bf16 As[2][128 * 32];
  __shared__ alignas(32) __bf16 Bs[2][128 * 32];
  const int m0 = blockIdx.x * 128;
  const int n0 = blockIdx.y * 128;
  const int t = threadIdx.x;                 // 256 threads = 8 waves
  const int wave = t >> 5, lane = t & 31;
  const int wm = wave & 3, wn = wave >> 2;   // 4 x 2 wave grid
  const int ln = lane & 15, g = lane >> 4;
  v8f z = {0.f,0.f,0.f,0.f,0.f,0.f,0.f,0.f};
  v8f acc00 = z, acc01 = z, acc02 = z, acc03 = z;   // M-subtile 0 (rows wm*32+0..15)
  v8f acc10 = z, acc11 = z, acc12 = z, acc13 = z;   // M-subtile 1 (rows wm*32+16..31)
  const int row_l = t >> 1;                  // 128 rows x 32 cols, 32B/thread
  const int ch_l  = (t & 1) * 16;

  const __bf16* gA  = &A[(size_t)(m0 + row_l) * 1024 + ch_l];
  const __bf16* gB  = &BT[(size_t)(n0 + row_l) * 1024 + ch_l];
  __bf16* lA0 = &As[0][row_l * 32 + ch_l];
  __bf16* lB0 = &Bs[0][row_l * 32 + ch_l];
  const int lstride = 128 * 32;              // buffer stride (halves)

#ifdef HAVE_ASYNC_LDS
  // prologue: DMA tile 0 into buffer 0 (2 x 16B per thread per matrix)
  async_copy16(gA, lA0);
  async_copy16(gA + 8, lA0 + 8);
  async_copy16(gB, lB0);
  async_copy16(gB + 8, lB0 + 8);
#endif

  for (int k0 = 0; k0 < 1024; k0 += 32) {
    const int buf = (k0 >> 5) & 1;
#ifdef HAVE_ASYNC_LDS
    wait_async0();                 // our DMAs into `buf` are done
    __syncthreads();               // everyone's DMAs done; no wave reads buf^1
    if (k0 + 32 < 1024) {          // overlap: DMA next tile while computing
      async_copy16(gA + k0 + 32, lA0 + (buf ^ 1) * lstride);
      async_copy16(gA + k0 + 40, lA0 + (buf ^ 1) * lstride + 8);
      async_copy16(gB + k0 + 32, lB0 + (buf ^ 1) * lstride);
      async_copy16(gB + k0 + 40, lB0 + (buf ^ 1) * lstride + 8);
    }
#else
    // fallback: batch loads into regs (one wait), then store to LDS
    v8bf ra0 = *(const v8bf*)(gA + k0);
    v8bf ra1 = *(const v8bf*)(gA + k0 + 8);
    v8bf rb0 = *(const v8bf*)(gB + k0);
    v8bf rb1 = *(const v8bf*)(gB + k0 + 8);
    __syncthreads();
    *(v8bf*)(lA0 + buf * lstride)     = ra0;
    *(v8bf*)(lA0 + buf * lstride + 8) = ra1;
    *(v8bf*)(lB0 + buf * lstride)     = rb0;
    *(v8bf*)(lB0 + buf * lstride + 8) = rb1;
    __syncthreads();
#endif
    if (k0 + 32 < 1024) {
      __builtin_prefetch(gA + k0 + 64, 0, 1);
      __builtin_prefetch(gB + k0 + 64, 0, 1);
    }
    const __bf16* Ab = &As[buf][0];
    const __bf16* Bb = &Bs[buf][0];
    const int arow0 = wm * 32 + ln;
    const int arow1 = arow0 + 16;
    v16bf a0 = *(const v16bf*)&Ab[arow0 * 32 + g * 16];
    v16bf a1 = *(const v16bf*)&Ab[arow1 * 32 + g * 16];
    v16bf b0 = *(const v16bf*)&Bb[(wn * 64 +  0 + ln) * 32 + g * 16];
    v16bf b1 = *(const v16bf*)&Bb[(wn * 64 + 16 + ln) * 32 + g * 16];
    v16bf b2 = *(const v16bf*)&Bb[(wn * 64 + 32 + ln) * 32 + g * 16];
    v16bf b3 = *(const v16bf*)&Bb[(wn * 64 + 48 + ln) * 32 + g * 16];
    acc00 = wmma_bf16(a0, b0, acc00);
    acc01 = wmma_bf16(a0, b1, acc01);
    acc02 = wmma_bf16(a0, b2, acc02);
    acc03 = wmma_bf16(a0, b3, acc03);
    acc10 = wmma_bf16(a1, b0, acc10);
    acc11 = wmma_bf16(a1, b1, acc11);
    acc12 = wmma_bf16(a1, b2, acc12);
    acc13 = wmma_bf16(a1, b3, acc13);
  }

  v8f accs[2][4] = {{acc00, acc01, acc02, acc03}, {acc10, acc11, acc12, acc13}};
  #pragma unroll
  for (int sub = 0; sub < 2; ++sub) {
    #pragma unroll
    for (int c = 0; c < 4; ++c) {
      #pragma unroll
      for (int r = 0; r < 8; ++r) {
        const int row = m0 + wm * 32 + sub * 16 + r + 8 * g;
        const int col = n0 + wn * 64 + c * 16 + ln;
        const float vv = accs[sub][c][r];
        if (mode == 0) {
          outb[(size_t)row * 1024 + col] = (__bf16)vv;
        } else if (mode == 1) {
          outb[((size_t)(row >> 11) * 1024 + col) * 2048 + (row & 2047)] = (__bf16)vv;
        } else {
          outf[(size_t)row * 1024 + col] = resid[(size_t)row * 1024 + col] + vv;
        }
      }
    }
  }
}

// ---------------- Flash attention with T5 bias (1 wave / 16 q-rows) ----------
__global__ void k_attention(const __bf16* __restrict__ Q, const __bf16* __restrict__ K,
                            const __bf16* __restrict__ VT, const float* __restrict__ blut,
                            __bf16* __restrict__ ctx) {
  __shared__ alignas(16) __bf16 Pl[16 * 32];
  const int qt = blockIdx.x & 127;
  const int bh = blockIdx.x >> 7;
  const int b = bh >> 4, h = bh & 15;
  const int lane = threadIdx.x, ln = lane & 15, g = lane >> 4;
  const int qbase = qt * 16;

  // Q A-operands (16x64 -> two K=32 operands), loop-invariant
  const __bf16* Qp = Q + (size_t)(b * S_ + qbase + ln) * HD_ + h * DK_;
  v16bf a0 = *(const v16bf*)&Qp[g * 16];
  v16bf a1 = *(const v16bf*)&Qp[32 + g * 16];

  float m_[8], l_[8];
  #pragma unroll
  for (int r = 0; r < 8; ++r) { m_[r] = -3.0e38f; l_[r] = 0.f; }
  v8f o0 = {0.f,0.f,0.f,0.f,0.f,0.f,0.f,0.f};
  v8f o1 = o0, o2 = o0, o3 = o0;

  const float* bl = blut + h * RELSPAN + (S_ - 1);
  const __bf16* Kb = K + (size_t)(b * S_) * HD_ + h * DK_;
  const __bf16* Vb = VT + ((size_t)b * HD_ + h * DK_) * (size_t)S_;

  for (int kt = 0; kt < S_ / 32; ++kt) {
    const int kb = kt * 32;
    // scores: B-operand rows = K[kpos][dk] (contiguous along dk)
    const __bf16* Kp0 = Kb + (size_t)(kb + ln) * HD_;
    const __bf16* Kp1 = Kp0 + (size_t)16 * HD_;
    v16bf bk00 = *(const v16bf*)&Kp0[g * 16];        // cols kb..+15, dk 0..31
    v16bf bk01 = *(const v16bf*)&Kp0[32 + g * 16];   // cols kb..+15, dk 32..63
    v16bf bk10 = *(const v16bf*)&Kp1[g * 16];        // cols kb+16..+31
    v16bf bk11 = *(const v16bf*)&Kp1[32 + g * 16];
    v8f s0 = {0.f,0.f,0.f,0.f,0.f,0.f,0.f,0.f};
    v8f s1 = s0;
    s0 = wmma_bf16(a0, bk00, s0);
    s0 = wmma_bf16(a1, bk01, s0);
    s1 = wmma_bf16(a0, bk10, s1);
    s1 = wmma_bf16(a1, bk11, s1);

    // bias + online softmax update; C-layout: row = r + 8*g, col = ln
    float scl[8];
    const int col0 = kb + ln, col1 = kb + 16 + ln;
    #pragma unroll
    for (int r = 0; r < 8; ++r) {
      const int q = qbase + r + 8 * g;
      float x0 = s0[r] + bl[col0 - q];
      float x1 = s1[r] + bl[col1 - q];
      float mx = rmax16(fmaxf(x0, x1));
      float mn = fmaxf(m_[r], mx);
      float sc = __expf(m_[r] - mn);
      m_[r] = mn;
      float p0 = __expf(x0 - mn);
      float p1 = __expf(x1 - mn);
      l_[r] = l_[r] * sc + rsum16(p0 + p1);
      scl[r] = sc;
      const int row = r + 8 * g;
      Pl[row * 32 + ln]      = (__bf16)p0;
      Pl[row * 32 + 16 + ln] = (__bf16)p1;
    }
    #pragma unroll
    for (int r = 0; r < 8; ++r) {
      o0[r] *= scl[r]; o1[r] *= scl[r]; o2[r] *= scl[r]; o3[r] *= scl[r];
    }
    // P as A-operand (LDS transpose bounce), V^T as B-operand (contiguous)
    v16bf aP = cat8(*(const v8bf*)&Pl[ln * 32 + g * 16],
                    *(const v8bf*)&Pl[ln * 32 + g * 16 + 8]);
    const __bf16* Vp = Vb + (size_t)ln * S_ + kb + g * 16;
    v16bf bv0 = *(const v16bf*)&Vp[0];
    v16bf bv1 = *(const v16bf*)&Vp[(size_t)16 * S_];
    v16bf bv2 = *(const v16bf*)&Vp[(size_t)32 * S_];
    v16bf bv3 = *(const v16bf*)&Vp[(size_t)48 * S_];
    o0 = wmma_bf16(aP, bv0, o0);
    o1 = wmma_bf16(aP, bv1, o1);
    o2 = wmma_bf16(aP, bv2, o2);
    o3 = wmma_bf16(aP, bv3, o3);
  }

  v8f oo[4] = {o0, o1, o2, o3};
  #pragma unroll
  for (int c = 0; c < 4; ++c) {
    #pragma unroll
    for (int r = 0; r < 8; ++r) {
      const int row = qbase + r + 8 * g;
      const int col = h * DK_ + c * 16 + ln;
      ctx[(size_t)(b * S_ + row) * HD_ + col] = (__bf16)(oo[c][r] / l_[r]);
    }
  }
}

// ---------------- launcher ---------------------------------------------------
extern "C" void kernel_launch(void* const* d_in, const int* in_sizes, int n_in,
                              void* d_out, int out_size, void* d_ws, size_t ws_size,
                              hipStream_t stream) {
  (void)in_sizes; (void)n_in; (void)out_size; (void)ws_size;
  const float* hidden   = (const float*)d_in[0];
  const float* ln_g     = (const float*)d_in[1];
  const float* ln_b     = (const float*)d_in[2];
  const float* Wq       = (const float*)d_in[3];
  const float* Wk       = (const float*)d_in[4];
  const float* Wv       = (const float*)d_in[5];
  const float* Wo       = (const float*)d_in[6];
  const float* rel_bias = (const float*)d_in[7];
  float* out = (float*)d_out;

  char* ws = (char*)d_ws;
  const size_t MB = 1024 * 1024;
  __bf16* normed = (__bf16*)(ws);             // 8 MB
  __bf16* wqT    = (__bf16*)(ws + 8  * MB);   // 2 MB each
  __bf16* wkT    = (__bf16*)(ws + 10 * MB);
  __bf16* wvT    = (__bf16*)(ws + 12 * MB);
  __bf16* woT    = (__bf16*)(ws + 14 * MB);
  __bf16* Qb     = (__bf16*)(ws + 16 * MB);   // 8 MB
  __bf16* Kbuf   = (__bf16*)(ws + 24 * MB);   // 8 MB
  __bf16* VTb    = (__bf16*)(ws + 32 * MB);   // 8 MB
  __bf16* ctx    = (__bf16*)(ws + 40 * MB);   // 8 MB
  float*  blut   = (float*)(ws + 48 * MB);    // 256 KB

  k_wT<<<4096, 256, 0, stream>>>(Wq, wqT);
  k_wT<<<4096, 256, 0, stream>>>(Wk, wkT);
  k_wT<<<4096, 256, 0, stream>>>(Wv, wvT);
  k_wT<<<4096, 256, 0, stream>>>(Wo, woT);
  k_bias_lut<<<(H_ * RELSPAN + 255) / 256, 256, 0, stream>>>(rel_bias, blut);
  k_layernorm<<<B_ * S_, 256, 0, stream>>>(hidden, ln_g, ln_b, normed);

  dim3 gg(32, 8);  // M=4096 / 128, N=1024 / 128
  k_gemm<<<gg, 256, 0, stream>>>(normed, wqT, 0, Qb,  nullptr, nullptr);
  k_gemm<<<gg, 256, 0, stream>>>(normed, wkT, 0, Kbuf, nullptr, nullptr);
  k_gemm<<<gg, 256, 0, stream>>>(normed, wvT, 1, VTb, nullptr, nullptr);

  k_attention<<<B_ * H_ * (S_ / 16), 32, 0, stream>>>(Qb, Kbuf, VTb, blut, ctx);

  k_gemm<<<gg, 256, 0, stream>>>(ctx, woT, 2, nullptr, out, hidden);
}
